// RSSM_31181462569322
// MI455X (gfx1250) — matrix operand order repaired and checked by reading, not compiled, not run on hardware
//
#include <hip/hip_runtime.h>
#include <hip/hip_bf16.h>

typedef __attribute__((ext_vector_type(16))) _Float16 v16h;
typedef __attribute__((ext_vector_type(8)))  _Float16 v8h;
typedef __attribute__((ext_vector_type(8)))  float    v8f;

union V16 { v16h v; v8h h[2]; };

#define BB   512
#define TT   512
#define OD   32
#define AD   8
#define NW   503   // T - W_LEN + 1
#define WLEN 10

// d_ws layout (in _Float16 units), all 16B aligned:
#define WS_W1T   0        // enc W1^T  [64][416] (zero-padded K)
#define WS_W2TE  26624    // enc W2^T  [64][64]
#define WS_W1TD  30720    // dec W1^T  [64][32]  (zero-padded K)
#define WS_W2TD  32768    // dec W2^T  [64][64]
#define WS_W3TD  36864    // dec W3^T  [32][64]
#define WS_TOTAL 38912    // halves (77,824 bytes)

// ---------------------------------------------------------------------------
// Prep: transpose + f16-convert weights into workspace (runs once per launch)
// ---------------------------------------------------------------------------
__global__ void rssm_prep(const float* __restrict__ eW1, const float* __restrict__ eW2,
                          const float* __restrict__ dW1, const float* __restrict__ dW2,
                          const float* __restrict__ dW3, _Float16* __restrict__ ws)
{
  const int tid = blockIdx.x * blockDim.x + threadIdx.x;
  const int stride = gridDim.x * blockDim.x;
  for (int i = tid; i < 64 * 416; i += stride) {
    int n = i / 416, k = i % 416;
    ws[WS_W1T + i] = (_Float16)(k < 400 ? eW1[k * 64 + n] : 0.f);
  }
  for (int i = tid; i < 64 * 64; i += stride) {
    int n = i / 64, k = i % 64;
    ws[WS_W2TE + i] = (_Float16)eW2[k * 64 + n];
    ws[WS_W2TD + i] = (_Float16)dW2[k * 64 + n];
  }
  for (int i = tid; i < 64 * 32; i += stride) {
    int n = i / 32, k = i % 32;
    ws[WS_W1TD + i] = (_Float16)(k < 10 ? dW1[k * 64 + n] : 0.f);
  }
  for (int i = tid; i < 32 * 64; i += stride) {
    int n = i / 64, k = i % 64;
    ws[WS_W3TD + i] = (_Float16)dW3[k * 32 + n];
  }
}

// A fragment (16-bit A 16x32, ISA 7.12.2): lane%16 = M, halves 0..7 hold
// K = kc+8*lg .. +7, halves 8..15 hold K = kc+16+8*lg .. +7  -> two b128 loads.
static __device__ __forceinline__ v16h load_afrag(const _Float16* rowp, int kc, int lg) {
  V16 A;
  A.h[0] = *(const v8h*)(rowp + kc + 8 * lg);
  A.h[1] = *(const v8h*)(rowp + kc + 16 + 8 * lg);
  return A.v;
}
// B fragment (16-bit B 32x16): lane%16 = N, halves 0..7 hold
// K = kc+8*lg .. +7, halves 8..15 hold K = kc+16+8*lg .. +7 -> two b128 loads from K-major W^T.
static __device__ __forceinline__ v16h load_bfrag(const _Float16* colp, int kc, int lg) {
  V16 Bv;
  Bv.h[0] = *(const v8h*)(colp + kc + 8 * lg);
  Bv.h[1] = *(const v8h*)(colp + kc + 16 + 8 * lg);
  return Bv.v;
}

// ---------------------------------------------------------------------------
// Encoder: priors = MLP3(win[B,N,400]; 400->64->64->2)
// ---------------------------------------------------------------------------
__global__ __launch_bounds__(256) void rssm_enc(
    const float* __restrict__ obs, const float* __restrict__ act,
    const _Float16* __restrict__ ws,
    const float* __restrict__ b1, const float* __restrict__ b2,
    const float* __restrict__ W3, const float* __restrict__ b3,
    float* __restrict__ priors)
{
  __shared__ __align__(16) float    oaL[41][40];
  __shared__ __align__(16) _Float16 xL[32][416];
  __shared__ __align__(16) _Float16 h1L[32][64];
  __shared__ __align__(16) _Float16 h2L[32][64];

  const int b   = blockIdx.y;
  const int n0  = blockIdx.x * 32;
  const int tid = threadIdx.x;

  // Stage the 41-timestep x 40-feature oa slice this block needs.
  for (int i = tid; i < 41 * 40; i += 256) {
    int tl = i / 40, f = i % 40;
    int tg = n0 + tl; if (tg > TT - 1) tg = TT - 1;
    oaL[tl][f] = (f < OD) ? obs[((size_t)b * TT + tg) * OD + f]
                          : act[((size_t)b * TT + tg) * AD + (f - OD)];
  }
  __syncthreads();

  // Expand window rows into f16 X tile: xL[r][f*10+w] = oaL[r+w][f], pad K 400..415.
  for (int i = tid; i < 32 * 400; i += 256) {
    int r = i / 400, k = i % 400;
    int f = k / WLEN, w = k % WLEN;
    xL[r][k] = (_Float16)oaL[r + w][f];
  }
  for (int i = tid; i < 32 * 16; i += 256)
    xL[i >> 4][400 + (i & 15)] = (_Float16)0.f;
  __syncthreads();

  const int lane = tid & 31;
  const int wave = tid >> 5;
  const int rT = wave >> 2;          // 0..1 : 16-row tile
  const int cT = wave & 3;           // 0..3 : 16-col tile
  const int lg = lane >> 4;
  const int m  = lane & 15;
  const int row = rT * 16 + m;
  const int nc  = cT * 16 + m;

  // ---- Layer 1: K = 416 (13 chunks of 32) ----
  const _Float16* w1p = ws + WS_W1T + (size_t)nc * 416;
  v8f c1 = {};
#pragma unroll
  for (int kc = 0; kc < 416; kc += 32) {
    v16h av = load_afrag(&xL[row][0], kc, lg);
    v16h bv = load_bfrag(w1p, kc, lg);
    c1 = __builtin_amdgcn_wmma_f32_16x16x32_f16(false, av, false, bv,
                                                (short)0, c1, false, false);
  }
  {
    float bb = b1[nc];
#pragma unroll
    for (int v = 0; v < 8; ++v)   // C/D layout: M = v + 8*lg, N = lane%16
      h1L[rT * 16 + v + 8 * lg][nc] = (_Float16)fmaxf(c1[v] + bb, 0.f);
  }
  __syncthreads();

  // ---- Layer 2: 64x64 ----
  const _Float16* w2p = ws + WS_W2TE + (size_t)nc * 64;
  v8f c2 = {};
#pragma unroll
  for (int kc = 0; kc < 64; kc += 32) {
    v16h av = load_afrag(&h1L[row][0], kc, lg);
    v16h bv = load_bfrag(w2p, kc, lg);
    c2 = __builtin_amdgcn_wmma_f32_16x16x32_f16(false, av, false, bv,
                                                (short)0, c2, false, false);
  }
  {
    float bb = b2[nc];
#pragma unroll
    for (int v = 0; v < 8; ++v)
      h2L[rT * 16 + v + 8 * lg][nc] = (_Float16)fmaxf(c2[v] + bb, 0.f);
  }
  __syncthreads();

  // ---- Layer 3: 64 -> 2 (tiny; scalar) ----
  if (tid < 64) {
    int r = tid >> 1, c = tid & 1;
    float s = b3[c];
#pragma unroll 8
    for (int k = 0; k < 64; ++k) s += (float)h2L[r][k] * W3[k * 2 + c];
    int n = n0 + r;
    if (n < NW) priors[((size_t)b * NW + n) * 2 + c] = s;
  }
}

// ---------------------------------------------------------------------------
// GRU scan: one thread per batch row, 502 sequential steps
// ---------------------------------------------------------------------------
__global__ __launch_bounds__(256) void rssm_scan(
    const float* __restrict__ priors, const float* __restrict__ act,
    const int* __restrict__ mask,
    const float* __restrict__ Wih, const float* __restrict__ Whh,
    const float* __restrict__ bih, const float* __restrict__ bhh,
    const float* __restrict__ pW1, const float* __restrict__ pb1,
    const float* __restrict__ pW2, const float* __restrict__ pb2,
    const float* __restrict__ pW3, const float* __restrict__ pb3,
    float* __restrict__ posts, float* __restrict__ xtraj)
{
  __shared__ float sw[438];
  const int tid = threadIdx.x;
  for (int i = tid; i < 438; i += 256) {
    float v;
    if      (i <  60) v = Wih[i];
    else if (i <  72) v = Whh[i - 60];
    else if (i <  78) v = bih[i - 72];
    else if (i <  84) v = bhh[i - 78];
    else if (i < 116) v = pW1[i - 84];
    else if (i < 132) v = pb1[i - 116];
    else if (i < 388) v = pW2[i - 132];
    else if (i < 404) v = pb2[i - 388];
    else if (i < 436) v = pW3[i - 404];
    else              v = pb3[i - 436];
    sw[i] = v;
  }
  __syncthreads();
  const float* sWih = sw;        const float* sWhh = sw + 60;
  const float* sbih = sw + 72;   const float* sbhh = sw + 78;
  const float* sW1  = sw + 84;   const float* sb1  = sw + 116;
  const float* sW2  = sw + 132;  const float* sb2  = sw + 388;
  const float* sW3  = sw + 404;  const float* sb3  = sw + 436;

  const int b = blockIdx.x * blockDim.x + tid;
  if (b >= BB) return;

  float x0 = 0.f, x1 = 0.f;
  float s0 = priors[((size_t)b * NW) * 2 + 0];
  float s1 = priors[((size_t)b * NW) * 2 + 1];
  posts[((size_t)b * NW) * 2 + 0] = s0;
  posts[((size_t)b * NW) * 2 + 1] = s1;
  xtraj[((size_t)b * NW) * 2 + 0] = 0.f;
  xtraj[((size_t)b * NW) * 2 + 1] = 0.f;

  for (int t = 0; t < NW - 1; ++t) {
    float mm = (float)mask[(size_t)b * NW + t];
    float p0 = priors[((size_t)b * NW + t) * 2 + 0];
    float p1 = priors[((size_t)b * NW + t) * 2 + 1];
    float st0 = mm * p0 + (1.f - mm) * s0;
    float st1 = mm * p1 + (1.f - mm) * s1;
    float gin[10];
#pragma unroll
    for (int i = 0; i < 8; ++i)
      gin[i] = act[((size_t)b * TT + (t + WLEN - 1)) * AD + i];
    gin[8] = st0; gin[9] = st1;
    float gi[6], gh[6];
#pragma unroll
    for (int j = 0; j < 6; ++j) {
      float a = sbih[j];
#pragma unroll
      for (int i = 0; i < 10; ++i) a += gin[i] * sWih[i * 6 + j];
      gi[j] = a;
      gh[j] = x0 * sWhh[j] + x1 * sWhh[6 + j] + sbhh[j];
    }
    float xn[2];
#pragma unroll
    for (int c = 0; c < 2; ++c) {
      float r = 1.f / (1.f + __expf(-(gi[c] + gh[c])));
      float z = 1.f / (1.f + __expf(-(gi[2 + c] + gh[2 + c])));
      float nn = tanhf(gi[4 + c] + r * gh[4 + c]);
      float xp = (c == 0) ? x0 : x1;
      xn[c] = (1.f - z) * nn + z * xp;
    }
    x0 = xn[0]; x1 = xn[1];
    float h1[16], h2[16];
#pragma unroll
    for (int j = 0; j < 16; ++j)
      h1[j] = fmaxf(x0 * sW1[j] + x1 * sW1[16 + j] + sb1[j], 0.f);
#pragma unroll
    for (int j = 0; j < 16; ++j) {
      float a = sb2[j];
#pragma unroll
      for (int i = 0; i < 16; ++i) a += h1[i] * sW2[i * 16 + j];
      h2[j] = fmaxf(a, 0.f);
    }
    s0 = sb3[0]; s1 = sb3[1];
#pragma unroll
    for (int i = 0; i < 16; ++i) {
      s0 += h2[i] * sW3[i * 2 + 0];
      s1 += h2[i] * sW3[i * 2 + 1];
    }
    posts[((size_t)b * NW + t + 1) * 2 + 0] = s0;
    posts[((size_t)b * NW + t + 1) * 2 + 1] = s1;
    xtraj[((size_t)b * NW + t + 1) * 2 + 0] = x0;
    xtraj[((size_t)b * NW + t + 1) * 2 + 1] = x1;
  }
}

// ---------------------------------------------------------------------------
// Decoder: pred = MLP3(concat(posts, action_traj); 10->64->64->32)
// ---------------------------------------------------------------------------
__global__ __launch_bounds__(256) void rssm_dec(
    const float* __restrict__ posts, const float* __restrict__ act,
    const _Float16* __restrict__ ws,
    const float* __restrict__ b1, const float* __restrict__ b2,
    const float* __restrict__ b3,
    float* __restrict__ pred)
{
  __shared__ __align__(16) _Float16 xdL[32][32];
  __shared__ __align__(16) _Float16 h1L[32][64];
  __shared__ __align__(16) _Float16 h2L[32][64];

  const int b   = blockIdx.y;
  const int n0  = blockIdx.x * 32;
  const int tid = threadIdx.x;

  // Stage padded input tile: [posts(2), action(8), 0...]
  for (int i = tid; i < 32 * 32; i += 256) {
    int r = i >> 5, k = i & 31;
    int n = n0 + r; if (n > NW - 1) n = NW - 1;
    float x = 0.f;
    if (k < 2)       x = posts[((size_t)b * NW + n) * 2 + k];
    else if (k < 10) x = act[((size_t)b * TT + (n + WLEN - 1)) * AD + (k - 2)];
    xdL[r][k] = (_Float16)x;
  }
  __syncthreads();

  const int lane = tid & 31, wave = tid >> 5;
  const int rT = wave >> 2, cT = wave & 3;
  const int lg = lane >> 4, m = lane & 15;
  const int row = rT * 16 + m;
  const int nc  = cT * 16 + m;

  // ---- Layer 1: K = 32 (padded), single WMMA ----
  v8f c1 = {};
  {
    v16h av = load_afrag(&xdL[row][0], 0, lg);
    v16h bv = load_bfrag(ws + WS_W1TD + (size_t)nc * 32, 0, lg);
    c1 = __builtin_amdgcn_wmma_f32_16x16x32_f16(false, av, false, bv,
                                                (short)0, c1, false, false);
  }
  {
    float bb = b1[nc];
#pragma unroll
    for (int v = 0; v < 8; ++v)
      h1L[rT * 16 + v + 8 * lg][nc] = (_Float16)fmaxf(c1[v] + bb, 0.f);
  }
  __syncthreads();

  // ---- Layer 2: 64x64 ----
  const _Float16* w2p = ws + WS_W2TD + (size_t)nc * 64;
  v8f c2 = {};
#pragma unroll
  for (int kc = 0; kc < 64; kc += 32) {
    v16h av = load_afrag(&h1L[row][0], kc, lg);
    v16h bv = load_bfrag(w2p, kc, lg);
    c2 = __builtin_amdgcn_wmma_f32_16x16x32_f16(false, av, false, bv,
                                                (short)0, c2, false, false);
  }
  {
    float bb = b2[nc];
#pragma unroll
    for (int v = 0; v < 8; ++v)
      h2L[rT * 16 + v + 8 * lg][nc] = (_Float16)fmaxf(c2[v] + bb, 0.f);
  }
  __syncthreads();

  // ---- Layer 3: 64 -> 32 (2x2 wave tiles, waves 0..3) ----
  if (wave < 4) {
    const int rT3 = wave >> 1, cT3 = wave & 1;
    const int row3 = rT3 * 16 + m;
    const int nc3  = cT3 * 16 + m;
    const _Float16* w3p = ws + WS_W3TD + (size_t)nc3 * 64;
    v8f c3 = {};
#pragma unroll
    for (int kc = 0; kc < 64; kc += 32) {
      v16h av = load_afrag(&h2L[row3][0], kc, lg);
      v16h bv = load_bfrag(w3p, kc, lg);
      c3 = __builtin_amdgcn_wmma_f32_16x16x32_f16(false, av, false, bv,
                                                  (short)0, c3, false, false);
    }
    float bb = b3[nc3];
#pragma unroll
    for (int v = 0; v < 8; ++v) {
      int r = rT3 * 16 + v + 8 * lg;
      int nn = n0 + r;
      if (nn < NW) pred[((size_t)b * NW + nn) * 32 + nc3] = c3[v] + bb;
    }
  }
}

// ---------------------------------------------------------------------------
// Pass-through outputs: target_obs_traj (float4 rows) and labels
// ---------------------------------------------------------------------------
__global__ void rssm_copy(const float* __restrict__ obs, const int* __restrict__ tl,
                          float* __restrict__ tgt, float* __restrict__ lab)
{
  const size_t stride = (size_t)gridDim.x * blockDim.x;
  const float4* src4 = (const float4*)obs;
  float4* dst4 = (float4*)tgt;
  const size_t tot1 = (size_t)BB * NW * (OD / 4);   // float4 elements
  for (size_t i = (size_t)blockIdx.x * blockDim.x + threadIdx.x; i < tot1; i += stride) {
    size_t b = i / ((size_t)NW * (OD / 4));
    size_t r = i % ((size_t)NW * (OD / 4));
    size_t n = r / (OD / 4);
    size_t c = r % (OD / 4);
    dst4[i] = src4[(b * TT + n + WLEN - 1) * (OD / 4) + c];
  }
  const size_t tot2 = (size_t)BB * NW;
  for (size_t i = (size_t)blockIdx.x * blockDim.x + threadIdx.x; i < tot2; i += stride) {
    size_t b = i / NW, n = i % NW;
    lab[i] = (float)tl[b * TT + n + WLEN - 1];
  }
}

extern "C" void kernel_launch(void* const* d_in, const int* in_sizes, int n_in,
                              void* d_out, int out_size, void* d_ws, size_t ws_size,
                              hipStream_t stream)
{
  (void)in_sizes; (void)n_in; (void)out_size; (void)ws_size;
  const float* obs  = (const float*)d_in[0];
  const float* act  = (const float*)d_in[1];
  const int*   tl   = (const int*)  d_in[2];
  const int*   mask = (const int*)  d_in[3];
  const float* eW1 = (const float*)d_in[4];   const float* eb1 = (const float*)d_in[5];
  const float* eW2 = (const float*)d_in[6];   const float* eb2 = (const float*)d_in[7];
  const float* eW3 = (const float*)d_in[8];   const float* eb3 = (const float*)d_in[9];
  const float* gWih = (const float*)d_in[10]; const float* gWhh = (const float*)d_in[11];
  const float* gbih = (const float*)d_in[12]; const float* gbhh = (const float*)d_in[13];
  const float* pW1 = (const float*)d_in[14];  const float* pb1 = (const float*)d_in[15];
  const float* pW2 = (const float*)d_in[16];  const float* pb2 = (const float*)d_in[17];
  const float* pW3 = (const float*)d_in[18];  const float* pb3 = (const float*)d_in[19];
  const float* dW1 = (const float*)d_in[20];  const float* db1 = (const float*)d_in[21];
  const float* dW2 = (const float*)d_in[22];  const float* db2 = (const float*)d_in[23];
  const float* dW3 = (const float*)d_in[24];  const float* db3 = (const float*)d_in[25];

  float* out = (float*)d_out;
  const size_t BN = (size_t)BB * NW;
  float* pred   = out;                 // (B,N,32)
  float* priors = pred  + BN * 32;     // (B,N,2)
  float* posts  = priors + BN * 2;     // (B,N,2)
  float* xtraj  = posts  + BN * 2;     // (B,N,2)
  float* tgt    = xtraj  + BN * 2;     // (B,N,32)
  float* lab    = tgt    + BN * 32;    // (B,N)

  _Float16* ws = (_Float16*)d_ws;

  rssm_prep<<<dim3(64), 256, 0, stream>>>(eW1, eW2, dW1, dW2, dW3, ws);

  dim3 gTile((NW + 31) / 32, BB);
  rssm_enc<<<gTile, 256, 0, stream>>>(obs, act, ws, eb1, eb2, eW3, eb3, priors);
  rssm_scan<<<dim3(2), 256, 0, stream>>>(priors, act, mask, gWih, gWhh, gbih, gbhh,
                                         pW1, pb1, pW2, pb2, pW3, pb3, posts, xtraj);
  rssm_dec<<<gTile, 256, 0, stream>>>(posts, act, ws, db1, db2, db3, pred);
  rssm_copy<<<dim3(512), 256, 0, stream>>>(obs, tl, tgt, lab);
}
